// TransformerBlockRPE_52355651338938
// MI455X (gfx1250) — matrix operand order repaired and checked by reading, not compiled, not run on hardware
//
#include <hip/hip_runtime.h>

// ---------------------------------------------------------------------------
// MI455X (gfx1250, wave32) implementation of the RPE-biased transformer block.
// Strategy: algebraically eliminate the [B,S,T,2D] (268MB) intermediate,
// run all dense GEMMs through v_wmma_f32_16x16x32_bf16 (fp32 accum),
// and do the two rpe-coupled skinny contractions as LDS-staged VALU kernels.
// GEMM wave tile is 16x32 (two accumulators sharing one A fragment) to cut
// A-side L2 traffic in half and double WMMA issue density.
// ---------------------------------------------------------------------------

typedef __bf16 bf16;
typedef __bf16 v16bf __attribute__((ext_vector_type(16)));
typedef __bf16 v8bf  __attribute__((ext_vector_type(8)));
typedef float  v8f   __attribute__((ext_vector_type(8)));

static constexpr int Dm = 256;   // d_model
static constexpr int Hh = 8;     // heads
static constexpr int Dh = 32;    // head dim
static constexpr int DR = 64;    // d_rpe
static constexpr int Ss = 256;
static constexpr int Tt = 256;
#define LN_EPS 1e-5f

// ---------------------------------------------------------------------------
// Generic bf16 WMMA GEMM:  C[M,N] = A[M,K] @ B[K,N] (+bias)(+residual)(ReLU?)
// B is supplied K-major ("Bt", shape [N,K]) so B-fragment loads are contiguous.
// One wave computes a 16x32 tile (two 16x16 WMMA accumulators, shared A frag).
// Host launches ceil(N/32) waves total across (blockDim/32)*gridDim.y; coverage
// is exact (all N used are multiples of 32).
// Batched via grid.z with composite offsets: off = (z/zq)*O1 + (z%zq)*O2.
// ---------------------------------------------------------------------------
__global__ __launch_bounds__(256)
void gemm_bf16_wmma(const bf16* __restrict__ A, const bf16* __restrict__ Bt,
                    const float* __restrict__ bias, const float* __restrict__ resid,
                    float* __restrict__ Cf, bf16* __restrict__ Cb,
                    int M, int N, int K, int lda, int ldb, int ldc,
                    long long aO1, long long aO2, long long bO1, long long bO2,
                    long long cO1, long long cO2, int zq, int relu)
{
    int wave   = threadIdx.x >> 5;
    int lane   = threadIdx.x & 31;
    int hw     = lane >> 4;          // half-wave (0: lanes 0-15, 1: lanes 16-31)
    int nwaves = blockDim.x >> 5;
    int m0     = blockIdx.x * 16;
    int n0     = (blockIdx.y * nwaves + wave) * 32;
    int z = blockIdx.z;
    long long za = (long long)(z / zq) * aO1 + (long long)(z % zq) * aO2;
    long long zb = (long long)(z / zq) * bO1 + (long long)(z % zq) * bO2;
    long long zc = (long long)(z / zq) * cO1 + (long long)(z % zq) * cO2;

    const bf16* Arow  = A  + za + (long long)(m0 + (lane & 15)) * lda;
    const bf16* Brow0 = Bt + zb + (long long)(n0 + (lane & 15)) * ldb;
    const bf16* Brow1 = Brow0 + (long long)16 * ldb;

    v8f acc0 = {0.f, 0.f, 0.f, 0.f, 0.f, 0.f, 0.f, 0.f};
    v8f acc1 = {0.f, 0.f, 0.f, 0.f, 0.f, 0.f, 0.f, 0.f};
    for (int k0 = 0; k0 < K; k0 += 32) {
        union { v16bf v; v8bf h[2]; } af, b0, b1;
        // A 16x32 bf16 fragment: lane half 0 holds K {0..15}, half 1 {16..31}
        af.h[0] = *(const v8bf*)(Arow + k0 + hw * 16);
        af.h[1] = *(const v8bf*)(Arow + k0 + hw * 16 + 8);
        // B 32x16 fragments (K-major storage): half 0 K {0..15}, half 1 K {16..31}
        b0.h[0] = *(const v8bf*)(Brow0 + k0 + hw * 16);
        b0.h[1] = *(const v8bf*)(Brow0 + k0 + hw * 16 + 8);
        b1.h[0] = *(const v8bf*)(Brow1 + k0 + hw * 16);
        b1.h[1] = *(const v8bf*)(Brow1 + k0 + hw * 16 + 8);
        acc0 = __builtin_amdgcn_wmma_f32_16x16x32_bf16(
                   false, af.v, false, b0.v, (short)0, acc0, false, false);
        acc1 = __builtin_amdgcn_wmma_f32_16x16x32_bf16(
                   false, af.v, false, b1.v, (short)0, acc1, false, false);
    }
    int cn0 = n0 + (lane & 15);
    int cn1 = cn0 + 16;
    float bv0 = bias ? bias[cn0] : 0.f;
    float bv1 = bias ? bias[cn1] : 0.f;
    #pragma unroll
    for (int r = 0; r < 8; ++r) {
        int cm = m0 + r + 8 * hw;                 // C/D layout: VGPR r -> row r + 8*half
        long long rowb = zc + (long long)cm * ldc;
        float v0 = acc0[r] + bv0;
        float v1 = acc1[r] + bv1;
        if (resid) { v0 += resid[rowb + cn0]; v1 += resid[rowb + cn1]; }
        if (relu)  { v0 = fmaxf(v0, 0.f); v1 = fmaxf(v1, 0.f); }
        if (Cf) { Cf[rowb + cn0] = v0; Cf[rowb + cn1] = v1; }
        if (Cb) { Cb[rowb + cn0] = (bf16)v0; Cb[rowb + cn1] = (bf16)v1; }
    }
}

// ---------------------------------------------------------------------------
// LayerNorm over D=256, one block per row, output bf16 for WMMA consumption.
// ---------------------------------------------------------------------------
__global__ __launch_bounds__(256)
void ln_to_bf16(const float* __restrict__ x, const float* __restrict__ g,
                const float* __restrict__ b, bf16* __restrict__ out)
{
    __shared__ float red[256];
    int row = blockIdx.x, tid = threadIdx.x;
    float v = x[(long long)row * Dm + tid];
    red[tid] = v; __syncthreads();
    for (int s = 128; s > 0; s >>= 1) { if (tid < s) red[tid] += red[tid + s]; __syncthreads(); }
    float m = red[0] * (1.f / Dm); __syncthreads();
    float d = v - m;
    red[tid] = d * d; __syncthreads();
    for (int s = 128; s > 0; s >>= 1) { if (tid < s) red[tid] += red[tid + s]; __syncthreads(); }
    float var = red[0] * (1.f / Dm);
    float nv = d * rsqrtf(var + LN_EPS) * g[tid] + b[tid];
    out[(long long)row * Dm + tid] = (bf16)nv;
}

// ---------------------------------------------------------------------------
// Fused: scores = (q.k + rpe.qw)*scale, mask, softmax -> w (fp32 in place + bf16)
// One block per (b,h,s) row; thread = t.
// ---------------------------------------------------------------------------
__global__ __launch_bounds__(256)
void score_softmax(float* __restrict__ scores, const float* __restrict__ rpe,
                   const float* __restrict__ qw, const unsigned char* __restrict__ tmask,
                   bf16* __restrict__ wbf)
{
    __shared__ float sqw[DR];
    __shared__ float red[256];
    int blk = blockIdx.x;
    int s = blk & (Ss - 1);
    int h = (blk >> 8) & (Hh - 1);
    int b = blk >> 11;
    int t = threadIdx.x;
    if (t < DR) sqw[t] = qw[((long long)(b * Ss + s)) * (Hh * DR) + h * DR + t];
    __syncthreads();

    long long sidx = (((long long)(b * Hh + h)) * Ss + s) * Tt + t;
    float val = scores[sidx];
    const float* rr = rpe + (((long long)(b * Ss + s)) * Tt + t) * DR;
    float dot = 0.f;
    #pragma unroll 8
    for (int j = 0; j < DR; ++j) dot += rr[j] * sqw[j];
    const float scale = 0.17677669529663687f;    // 1/sqrt(32)
    val = (val + dot) * scale;
    if (tmask[b * Tt + t]) val = -3.0e38f;       // -inf surrogate

    red[t] = val; __syncthreads();
    for (int k = 128; k > 0; k >>= 1) { if (t < k) red[t] = fmaxf(red[t], red[t + k]); __syncthreads(); }
    float mx = red[0]; __syncthreads();
    float ex = __expf(val - mx);
    red[t] = ex; __syncthreads();
    for (int k = 128; k > 0; k >>= 1) { if (t < k) red[t] += red[t + k]; __syncthreads(); }
    float w = ex / red[0];
    scores[sidx] = w;
    wbf[sidx] = (bf16)w;
}

// v [b,t,h,d] (bf16) -> vT [b,h,d,t] (bf16), K-major operand for the w.v GEMM
__global__ __launch_bounds__(256)
void transpose_v_kernel(const bf16* __restrict__ v, bf16* __restrict__ vT)
{
    long long idx = (long long)blockIdx.x * 256 + threadIdx.x;  // [b][h][d][t]
    int t = idx & (Tt - 1);
    int d = (idx >> 8) & (Dh - 1);
    int h = (idx >> 13) & (Hh - 1);
    int b = (int)(idx >> 16);
    vT[idx] = v[((long long)(b * Tt + t)) * Dm + h * Dh + d];
}

// ---------------------------------------------------------------------------
// p[h,j] = sum_t w[b,h,s,t]*rpe[b,s,t,j]; out = attnv + p@wr_v + br_v (bf16)
// One block per (b,s). LDS-staged rpe chunks (64 t x 64 j) + w rows.
// ---------------------------------------------------------------------------
__global__ __launch_bounds__(256)
void pv_assemble(const float* __restrict__ w, const float* __restrict__ rpe,
                 const float* __restrict__ attnv, const float* __restrict__ wr,
                 const float* __restrict__ br, bf16* __restrict__ outbf)
{
    __shared__ float sw[Hh][Tt];        // 8 KB
    __shared__ float srpe[64][DR + 1];  // ~16.6 KB (padded vs bank conflicts)
    __shared__ float sp[Hh][DR];        // 2 KB
    int blk = blockIdx.x;
    int s = blk & (Ss - 1);
    int b = blk >> 8;
    int tid = threadIdx.x;

    #pragma unroll
    for (int r = 0; r < 8; ++r) {
        int lin = r * 256 + tid; int h = lin >> 8; int t = lin & 255;
        sw[h][t] = w[(((long long)(b * Hh + h)) * Ss + s) * Tt + t];
    }

    int h0 = tid >> 6, j0 = tid & 63;   // pair 0 = tid, pair 1 = tid+256 (same j)
    int h1 = h0 + 4;
    float acc0 = 0.f, acc1 = 0.f;
    for (int tc = 0; tc < Tt; tc += 64) {
        __syncthreads();
        #pragma unroll
        for (int r = 0; r < 16; ++r) {
            int lin = r * 256 + tid; int tt = lin >> 6; int jj = lin & 63;
            srpe[tt][jj] = rpe[(((long long)(b * Ss + s)) * Tt + tc + tt) * DR + jj];
        }
        __syncthreads();
        for (int tt = 0; tt < 64; ++tt) {
            float rv = srpe[tt][j0];
            acc0 += sw[h0][tc + tt] * rv;
            acc1 += sw[h1][tc + tt] * rv;
        }
    }
    sp[h0][j0] = acc0;
    sp[h1][j0] = acc1;
    __syncthreads();

    int h = tid >> 5, d = tid & 31, col = h * Dh + d;
    float val = attnv[((long long)(b * Ss + s)) * Dm + col] + br[Dm + col];
    for (int j = 0; j < DR; ++j)
        val += sp[h][j] * wr[(long long)j * (2 * Dm) + Dm + col];
    outbf[((long long)(b * Ss + s)) * Dm + col] = (bf16)val;
}

// -------------------------- small utility kernels ---------------------------
__global__ void copy_f32_k(const float* __restrict__ in, float* __restrict__ out, int n)
{ int i = blockIdx.x * 256 + threadIdx.x; if (i < n) out[i] = in[i]; }

__global__ void mask_rows_k(float* __restrict__ buf, const unsigned char* __restrict__ m)
{ int i = blockIdx.x * 256 + threadIdx.x; if (m[i >> 8]) buf[i] = 0.f; }

__global__ void convert_bf16_k(const float* __restrict__ in, bf16* __restrict__ out, int n)
{ int i = blockIdx.x * 256 + threadIdx.x; if (i < n) out[i] = (bf16)in[i]; }

// WT[n*K + k] = (bf16) W[k*N + n]  (fp32 [K,N] -> K-major bf16 [N,K])
__global__ void convertT_bf16_k(const float* __restrict__ W, bf16* __restrict__ WT, int K, int N)
{
    long long i = (long long)blockIdx.x * 256 + threadIdx.x;
    if (i >= (long long)K * N) return;
    int k = (int)(i % K); int n = (int)(i / K);
    WT[i] = (bf16)W[(long long)k * N + n];
}

// ---------------------------------------------------------------------------
extern "C" void kernel_launch(void* const* d_in, const int* in_sizes, int n_in,
                              void* d_out, int out_size, void* d_ws, size_t ws_size,
                              hipStream_t stream)
{
    (void)in_sizes; (void)n_in; (void)out_size; (void)ws_size;
    const float* src  = (const float*)d_in[0];
    const float* tgt  = (const float*)d_in[1];
    const float* rpe  = (const float*)d_in[2];
    const unsigned char* smask = (const unsigned char*)d_in[3];  // numpy bool = u8
    const unsigned char* tmask = (const unsigned char*)d_in[4];
    const float* ln1_g = (const float*)d_in[5];
    const float* ln1_b = (const float*)d_in[6];
    const float* lnt_g = (const float*)d_in[7];
    const float* lnt_b = (const float*)d_in[8];
    const float* ln2_g = (const float*)d_in[9];
    const float* ln2_b = (const float*)d_in[10];

    char* ws = (char*)d_ws;
    size_t off = 0;
    auto alloc = [&](size_t bytes) -> void* {
        off = (off + 255) & ~(size_t)255;
        void* p = ws + off; off += bytes; return p;
    };
    float* srcbuf = (float*)alloc((size_t)512 * 256 * 4);
    float* qwbuf  = (float*)alloc((size_t)512 * 512 * 4);
    float* scores = (float*)alloc((size_t)2 * 8 * 256 * 256 * 4);
    float* attnv  = (float*)alloc((size_t)512 * 256 * 4);
    bf16* s2bf   = (bf16*)alloc((size_t)512 * 256 * 2);
    bf16* tnbf   = (bf16*)alloc((size_t)512 * 256 * 2);
    bf16* qbf    = (bf16*)alloc((size_t)512 * 256 * 2);
    bf16* kbf    = (bf16*)alloc((size_t)512 * 256 * 2);
    bf16* vbf    = (bf16*)alloc((size_t)512 * 256 * 2);
    bf16* vTbf   = (bf16*)alloc((size_t)512 * 256 * 2);
    bf16* aoutbf = (bf16*)alloc((size_t)512 * 256 * 2);
    bf16* wbf    = (bf16*)alloc((size_t)2 * 8 * 256 * 256 * 2);
    bf16* wqT    = (bf16*)alloc((size_t)65536 * 2);
    bf16* wkT    = (bf16*)alloc((size_t)65536 * 2);
    bf16* wvT    = (bf16*)alloc((size_t)65536 * 2);
    bf16* woT    = (bf16*)alloc((size_t)65536 * 2);
    bf16* wrbf   = (bf16*)alloc((size_t)32768 * 2);
    bf16* w1T    = (bf16*)alloc((size_t)262144 * 2);
    bf16* w2T    = (bf16*)alloc((size_t)262144 * 2);
    bf16* h1bf   = (bf16*)alloc((size_t)512 * 1024 * 2);

    // Each wave covers a 16x32 tile; pick block/grid so column coverage is exact.
    auto gemm = [&](const bf16* A, const bf16* Bt, const float* bias, const float* resid,
                    float* Cf, bf16* Cb, int M, int N, int K, int lda, int ldb, int ldc,
                    long long aO1, long long aO2, long long bO1, long long bO2,
                    long long cO1, long long cO2, int Z, int zq, int relu) {
        int tiles32 = N / 32;
        int wavesPerBlock = tiles32 < 8 ? tiles32 : 8;
        dim3 grid(M / 16, tiles32 / wavesPerBlock, Z);
        dim3 block(wavesPerBlock * 32);
        gemm_bf16_wmma<<<grid, block, 0, stream>>>(
            A, Bt, bias, resid, Cf, Cb, M, N, K, lda, ldb, ldc,
            aO1, aO2, bO1, bO2, cO1, cO2, zq, relu);
    };

    copy_f32_k<<<512, 256, 0, stream>>>(src, srcbuf, 131072);

    for (int i = 0; i < 2; ++i) {
        const float* wq = (const float*)d_in[11] + (size_t)i * 65536;
        const float* bq = (const float*)d_in[12] + (size_t)i * 256;
        const float* wk = (const float*)d_in[13] + (size_t)i * 65536;
        const float* bk = (const float*)d_in[14] + (size_t)i * 256;
        const float* wv = (const float*)d_in[15] + (size_t)i * 65536;
        const float* bv = (const float*)d_in[16] + (size_t)i * 256;
        const float* wo = (const float*)d_in[17] + (size_t)i * 65536;
        const float* bo = (const float*)d_in[18] + (size_t)i * 256;
        const float* wr = (const float*)d_in[19] + (size_t)i * 32768;
        const float* br = (const float*)d_in[20] + (size_t)i * 512;
        const float* w1 = (const float*)d_in[21] + (size_t)i * 262144;
        const float* b1 = (const float*)d_in[22] + (size_t)i * 1024;
        const float* w2 = (const float*)d_in[23] + (size_t)i * 262144;
        const float* b2 = (const float*)d_in[24] + (size_t)i * 256;

        // weight conversion (to K-major bf16 for WMMA B operands)
        convertT_bf16_k<<<256, 256, 0, stream>>>(wq, wqT, 256, 256);
        convertT_bf16_k<<<256, 256, 0, stream>>>(wk, wkT, 256, 256);
        convertT_bf16_k<<<256, 256, 0, stream>>>(wv, wvT, 256, 256);
        convertT_bf16_k<<<256, 256, 0, stream>>>(wo, woT, 256, 256);
        convert_bf16_k<<<128, 256, 0, stream>>>(wr, wrbf, 32768);   // wr rows already K-major per head
        convertT_bf16_k<<<1024, 256, 0, stream>>>(w1, w1T, 256, 1024);
        convertT_bf16_k<<<1024, 256, 0, stream>>>(w2, w2T, 1024, 256);

        // LayerNorms
        ln_to_bf16<<<512, 256, 0, stream>>>(srcbuf, ln1_g + i * 256, ln1_b + i * 256, s2bf);
        ln_to_bf16<<<512, 256, 0, stream>>>(tgt,    lnt_g + i * 256, lnt_b + i * 256, tnbf);

        // q,k,v projections: [512,256] = [512,256] @ [256,256]
        gemm(s2bf, wqT, bq, nullptr, nullptr, qbf, 512, 256, 256, 256, 256, 256,
             0, 0, 0, 0, 0, 0, 1, 1, 0);
        gemm(tnbf, wkT, bk, nullptr, nullptr, kbf, 512, 256, 256, 256, 256, 256,
             0, 0, 0, 0, 0, 0, 1, 1, 0);
        gemm(tnbf, wvT, bv, nullptr, nullptr, vbf, 512, 256, 256, 256, 256, 256,
             0, 0, 0, 0, 0, 0, 1, 1, 0);

        // qw[b,s,h,j] = q_h . wr_k(h)  (batched over h: M=512,N=64,K=32)
        gemm(qbf, wrbf, nullptr, nullptr, qwbuf, nullptr, 512, 64, 32, 256, 512, 512,
             32, 0, 32, 0, 64, 0, 8, 1, 0);

        // raw scores: q.k^T batched over (b,h): M=S,N=T,K=Dh
        gemm(qbf, kbf, nullptr, nullptr, scores, nullptr, 256, 256, 32, 256, 256, 256,
             65536, 32, 65536, 32, 524288, 65536, 16, 8, 0);

        // + rpe.qw, scale, mask, softmax
        score_softmax<<<4096, 256, 0, stream>>>(scores, rpe, qwbuf, tmask, wbf);

        // w @ v  (batched over (b,h): M=S,N=Dh,K=T) — needs v K-major
        transpose_v_kernel<<<512, 256, 0, stream>>>(vbf, vTbf);
        gemm(wbf, vTbf, nullptr, nullptr, attnv, nullptr, 256, 32, 256, 256, 256, 256,
             524288, 65536, 65536, 8192, 65536, 32, 16, 8, 0);

        // p = w.rpe; out = attnv + p@wr_v + br_v
        pv_assemble<<<512, 256, 0, stream>>>(scores, rpe, attnv, wr, br, aoutbf);

        // output projection + residual
        gemm(aoutbf, woT, bo, srcbuf, srcbuf, nullptr, 512, 256, 256, 256, 256, 256,
             0, 0, 0, 0, 0, 0, 1, 1, 0);

        // FFN
        ln_to_bf16<<<512, 256, 0, stream>>>(srcbuf, ln2_g + i * 256, ln2_b + i * 256, s2bf);
        gemm(s2bf, w1T, b1, nullptr, nullptr, h1bf, 512, 1024, 256, 256, 256, 1024,
             0, 0, 0, 0, 0, 0, 1, 1, 1 /*relu*/);
        gemm(h1bf, w2T, b2, srcbuf, srcbuf, nullptr, 512, 256, 1024, 1024, 1024, 256,
             0, 0, 0, 0, 0, 0, 1, 1, 0);

        // zero padded src rows (reference does this every layer)
        mask_rows_k<<<512, 256, 0, stream>>>(srcbuf, smask);
    }

    copy_f32_k<<<512, 256, 0, stream>>>(srcbuf, (float*)d_out, 131072);
}
